// AbsolutePool_38697655337320
// MI455X (gfx1250) — matrix operand order repaired and checked by reading, not compiled, not run on hardware
//
#include <hip/hip_runtime.h>

typedef _Float16 v16h __attribute__((ext_vector_type(16)));
typedef float    v8f  __attribute__((ext_vector_type(8)));

#define A_DIM 512
#define B_DIM 512
#define V_DIM 32
#define D_DIM 512
#define TEMP_INV 0.2f   // 1/5.0

// block = 128 threads (4 waves); each wave computes a 16-row A tile, block = 64 rows.
// grid = (B_DIM, A_DIM/64)
__global__ __launch_bounds__(128)
void AbsolutePool_kernel(const float* __restrict__ text,
                         const float* __restrict__ video,
                         float* __restrict__ out)
{
    __shared__ __align__(32) _Float16 sVideo [V_DIM * D_DIM];  // [v][d] f16, 32 KB
    __shared__ __align__(32) _Float16 sVideoT[D_DIM * V_DIM];  // [d][v] f16, 32 KB
    __shared__ __align__(32) float    sScores[4][16][V_DIM];   // per-wave scores, 8 KB
    __shared__ __align__(32) _Float16 sW     [4][16][V_DIM];   // per-wave softmax weights, 4 KB

    const int b     = blockIdx.x;
    const int aBase = blockIdx.y * 64;
    const int t     = threadIdx.x;
    const int wave  = t >> 5;
    const int lane  = t & 31;

    // ---------- Phase 0: stage video[b] to LDS as f16, row-major + transposed ----------
    const float* vb = video + (size_t)b * (V_DIM * D_DIM);
    #pragma unroll 4
    for (int i = 0; i < (V_DIM * D_DIM) / 128; ++i) {
        const int e = i * 128 + t;            // coalesced across threads
        const int v = e >> 9;                 // e / 512
        const int d = e & (D_DIM - 1);
        const _Float16 h = (_Float16)vb[e];
        sVideo [v * D_DIM + d] = h;
        sVideoT[d * V_DIM + v] = h;
    }
    __syncthreads();

    const int halfSel = lane >> 4;            // 0: lanes 0-15, 1: lanes 16-31
    const int lane16  = lane & 15;
    const int aRow    = aBase + wave * 16 + lane16;  // A-matrix row this lane feeds

    // ---------- Phase 1: scores[16x32] = text_tile[16x512] x video[b]^T ----------
    // A frag: lane = M row; halfs 0-15 -> K = k0..k0+15,
    // with k0 = ks*32 + halfSel*16.
    // B frag: lane16 = N (= v), 16 contiguous K (= d) halfs starting ks*32 + halfSel*16.
    v8f acc0 = {};                            // v = 0..15
    v8f acc1 = {};                            // v = 16..31
    const float* trow = text + (size_t)aRow * D_DIM;
    for (int ks = 0; ks < D_DIM / 32; ++ks) {
        const int k0 = ks * 32 + halfSel * 16;
        v16h Af;
        #pragma unroll
        for (int h = 0; h < 8; ++h) Af[h]     = (_Float16)trow[k0 + h];
        #pragma unroll
        for (int h = 0; h < 8; ++h) Af[8 + h] = (_Float16)trow[k0 + 8 + h];

        const int dstart = ks * 32 + halfSel * 16;
        v16h B0 = *(const v16h*)&sVideo[(     lane16) * D_DIM + dstart];
        v16h B1 = *(const v16h*)&sVideo[(16 + lane16) * D_DIM + dstart];

        acc0 = __builtin_amdgcn_wmma_f32_16x16x32_f16(false, Af, false, B0,
                                                      (short)0, acc0, false, false);
        acc1 = __builtin_amdgcn_wmma_f32_16x16x32_f16(false, Af, false, B1,
                                                      (short)0, acc1, false, false);
    }

    // C/D layout: VGPR r holds M = r + 8*halfSel, N = lane16.
    #pragma unroll
    for (int r = 0; r < 8; ++r) {
        const int m = halfSel * 8 + r;
        sScores[wave][m][lane16]      = acc0[r];
        sScores[wave][m][16 + lane16] = acc1[r];
    }
    __syncthreads();

    // ---------- Phase 2: softmax over v (temperature 5) ----------
    if (lane < 16) {
        const float* srow = &sScores[wave][lane][0];
        float mx = srow[0];
        #pragma unroll
        for (int v = 1; v < V_DIM; ++v) mx = fmaxf(mx, srow[v]);
        float e[V_DIM];
        float sum = 0.f;
        #pragma unroll
        for (int v = 0; v < V_DIM; ++v) {
            e[v] = __expf((srow[v] - mx) * TEMP_INV);
            sum += e[v];
        }
        const float inv = 1.f / sum;
        #pragma unroll
        for (int v = 0; v < V_DIM; ++v)
            sW[wave][lane][v] = (_Float16)(e[v] * inv);
    }
    __syncthreads();

    // ---------- Phase 3: pooled[16x512] = W[16x32] x video[b][32x512] ----------
    // A frag = W: lane16 = M row, K = v, 16 contiguous v starting at halfSel*16.
    v16h Aw;
    {
        const _Float16* wrow = &sW[wave][lane16][0];
        const int kb = halfSel * 16;
        #pragma unroll
        for (int h = 0; h < 8; ++h) Aw[h]     = wrow[kb + h];
        #pragma unroll
        for (int h = 0; h < 8; ++h) Aw[8 + h] = wrow[kb + 8 + h];
    }

    for (int nt = 0; nt < D_DIM / 16; ++nt) {
        const int d = nt * 16 + lane16;       // N (= d) for this lane
        // B frag: 16 contiguous K (= v) halfs at sVideoT[d][halfSel*16 ..]
        v16h Bv = *(const v16h*)&sVideoT[d * V_DIM + halfSel * 16];
        v8f z = {};
        v8f c = __builtin_amdgcn_wmma_f32_16x16x32_f16(false, Aw, false, Bv,
                                                       (short)0, z, false, false);
        #pragma unroll
        for (int r = 0; r < 8; ++r) {
            const int m = halfSel * 8 + r;
            const int a = aBase + wave * 16 + m;
            out[((size_t)a * B_DIM + b) * D_DIM + d] = c[r];
        }
    }
}

extern "C" void kernel_launch(void* const* d_in, const int* in_sizes, int n_in,
                              void* d_out, int out_size, void* d_ws, size_t ws_size,
                              hipStream_t stream)
{
    (void)in_sizes; (void)n_in; (void)out_size; (void)d_ws; (void)ws_size;
    const float* text  = (const float*)d_in[0];   // [512, 512] f32
    const float* video = (const float*)d_in[1];   // [512, 32, 512] f32
    float* out = (float*)d_out;                   // [512, 512, 512] f32

    dim3 grid(B_DIM, A_DIM / 64, 1);
    dim3 block(128, 1, 1);
    hipLaunchKernelGGL(AbsolutePool_kernel, grid, block, 0, stream, text, video, out);
}
